// Attention_90812788506696
// MI455X (gfx1250) — matrix operand order repaired and checked by reading, not compiled, not run on hardware
//
#include <hip/hip_runtime.h>
#include <hip/hip_bf16.h>
#include <math.h>
#include <stdint.h>

// ---------------------------------------------------------------------------
// Problem constants (from reference)
// ---------------------------------------------------------------------------
#define B_     4
#define N_     2048
#define C_     1024
#define H_     16
#define D_     64
#define M_     (B_ * N_)      // 8192 rows total
#define EPS_   1e-7f
#define SCALE_ 0.125f         // 64^-0.5

typedef _Float16 half_t;
typedef __attribute__((ext_vector_type(16))) _Float16 v16h;
typedef __attribute__((ext_vector_type(8)))  _Float16 v8h;
typedef __attribute__((ext_vector_type(4)))  _Float16 v4h;
typedef __attribute__((ext_vector_type(8)))  float    v8f;

union V16 { v16h v; v8h h[2]; };

__device__ __forceinline__ v8f wmma16(v16h a, v16h b, v8f c) {
  return __builtin_amdgcn_wmma_f32_16x16x32_f16(false, a, false, b, (short)0, c,
                                                false, false);
}

// Async global->LDS copy (CDNA5 async path, tracked by ASYNCcnt).
// vdst = LDS byte address (low 32 bits of generic shared pointer), vaddr = 64b.
__device__ __forceinline__ void async_ld128(uint32_t lds_byte, const void* g) {
  asm volatile("global_load_async_to_lds_b128 %0, %1, off"
               :: "v"(lds_byte), "v"((unsigned long long)(uintptr_t)g)
               : "memory");
}
#define WAIT_ASYNC(n) asm volatile("s_wait_asynccnt " #n ::: "memory")

__device__ __forceinline__ uint32_t lds_addr(const void* p) {
  return (uint32_t)(uintptr_t)p;
}

// ---------------------------------------------------------------------------
// Intra-16-lane butterfly reductions (matches WMMA C/D row<->lane mapping).
// Prefer v_permlane16_b32 (VALU crosslane, keeps the LDS pipe free for the
// B-tile ds_load_b128 stream); fall back to ds_bpermute shuffles.
// ---------------------------------------------------------------------------
#if __has_builtin(__builtin_amdgcn_permlane16)
#define USE_PERMLANE16 1
#else
#define USE_PERMLANE16 0
#endif

#if USE_PERMLANE16
__device__ __forceinline__ float pl16(float v, unsigned lo, unsigned hi) {
  unsigned u = __float_as_uint(v);
  u = __builtin_amdgcn_permlane16(u, u, lo, hi, false, false);
  return __uint_as_float(u);
}
__device__ __forceinline__ float rmax16(float v) {
  v = fmaxf(v, pl16(v, 0x67452301u, 0xEFCDAB89u));  // xor 1
  v = fmaxf(v, pl16(v, 0x54761032u, 0xDCFE98BAu));  // xor 2
  v = fmaxf(v, pl16(v, 0x32107654u, 0xBA98FEDCu));  // xor 4
  v = fmaxf(v, pl16(v, 0xFEDCBA98u, 0x76543210u));  // xor 8
  return v;
}
__device__ __forceinline__ float rsum16(float v) {
  v += pl16(v, 0x67452301u, 0xEFCDAB89u);
  v += pl16(v, 0x54761032u, 0xDCFE98BAu);
  v += pl16(v, 0x32107654u, 0xBA98FEDCu);
  v += pl16(v, 0xFEDCBA98u, 0x76543210u);
  return v;
}
#else
__device__ __forceinline__ float rmax16(float v) {
#pragma unroll
  for (int m = 8; m >= 1; m >>= 1) v = fmaxf(v, __shfl_xor(v, m, 32));
  return v;
}
__device__ __forceinline__ float rsum16(float v) {
#pragma unroll
  for (int m = 8; m >= 1; m >>= 1) v += __shfl_xor(v, m, 32);
  return v;
}
#endif

// ---------------------------------------------------------------------------
// Kernel 1: per-row std (ddof=1) normalization + f16 conversion of xn and x
// ---------------------------------------------------------------------------
__global__ void __launch_bounds__(256)
norm_cvt(const float* __restrict__ x, const float* __restrict__ gamma,
         half_t* __restrict__ xh, half_t* __restrict__ xf) {
  __shared__ float ps[8], pq[8];
  __shared__ float sinv;
  const size_t row = blockIdx.x;
  const float4 xv = ((const float4*)(x + row * C_))[threadIdx.x];
  float s = xv.x + xv.y + xv.z + xv.w;
  float q = xv.x * xv.x + xv.y * xv.y + xv.z * xv.z + xv.w * xv.w;
#pragma unroll
  for (int m = 16; m >= 1; m >>= 1) {
    s += __shfl_xor(s, m, 32);
    q += __shfl_xor(q, m, 32);
  }
  if ((threadIdx.x & 31) == 0) { ps[threadIdx.x >> 5] = s; pq[threadIdx.x >> 5] = q; }
  __syncthreads();
  if (threadIdx.x == 0) {
    float S = 0.f, Q = 0.f;
#pragma unroll
    for (int i = 0; i < 8; ++i) { S += ps[i]; Q += pq[i]; }
    float mean = S * (1.f / C_);
    float var  = (Q - S * mean) * (1.f / (C_ - 1));
    sinv = 1.f / (sqrtf(var) + EPS_);
  }
  __syncthreads();
  const float inv = sinv;
  const float* g = gamma + threadIdx.x * 4;
  v4h on, ox;
  on[0] = (half_t)(xv.x * inv * g[0]);
  on[1] = (half_t)(xv.y * inv * g[1]);
  on[2] = (half_t)(xv.z * inv * g[2]);
  on[3] = (half_t)(xv.w * inv * g[3]);
  ox[0] = (half_t)xv.x; ox[1] = (half_t)xv.y;
  ox[2] = (half_t)xv.z; ox[3] = (half_t)xv.w;
  ((v4h*)(xh + row * C_))[threadIdx.x] = on;
  ((v4h*)(xf + row * C_))[threadIdx.x] = ox;
}

// ---------------------------------------------------------------------------
// Kernel 2: f32 -> f16 weight conversion
// ---------------------------------------------------------------------------
__global__ void f2h(const float* __restrict__ src, half_t* __restrict__ dst, int n) {
  int i = blockIdx.x * blockDim.x + threadIdx.x;
  if (i < n) dst[i] = (half_t)src[i];
}

// ---------------------------------------------------------------------------
// Kernel 3: Out[M,Nout] = In[M,K] * W[Nout,K]^T (v_wmma_f32_16x16x32_f16)
// 8 waves; wave w -> rows [by*256 + 32w, +32) as TWO 16-row tiles sharing the
// LDS B tiles, cols [bx*64,+64): 8 WMMA per k-step for 8 ds_load_b128.
// W tiles double-buffered via global_load_async_to_lds_b128 / s_wait_asynccnt.
// MODE 0: f16 out (scaleA). MODE 1: f32 out * colscale. MODE 2: KV split:
//   cols 0..63 -> kh[row*64+col]; cols 64..127 -> vt[(b*64+col-64)*N + n].
// ---------------------------------------------------------------------------
template <int MODE>
__global__ void __launch_bounds__(256)
gemm_rt(const half_t* __restrict__ A, const half_t* __restrict__ W,
        void* __restrict__ Out, half_t* __restrict__ out2,
        const float* __restrict__ colscale, int K, int Nout, float scaleA) {
  __shared__ __align__(16) half_t wt[2][64 * 32];
  const int tid = threadIdx.x;
  const int wv = tid >> 5, lane = tid & 31, li = lane & 15;
  const bool hi = lane >= 16;
  const int n0 = blockIdx.x * 64;
  const int m0 = blockIdx.y * 256 + wv * 32;
  v8f acc[2][4] = {};
  // staging: 64 rows x 32 halves = 4KB = 256 x 16B units, one per thread
  const int srow = tid >> 2, sseg = tid & 3;
  const uint32_t sdst = lds_addr(&wt[0][0]) + (uint32_t)(srow * 64 + sseg * 16);

  async_ld128(sdst, &W[(size_t)(n0 + srow) * K + sseg * 8]);
  int p = 0;
  for (int k0 = 0; k0 < K; k0 += 32) {
    __syncthreads();  // all waves done reading wt[1-p]
    const bool more = (k0 + 32) < K;
    if (more) {
      async_ld128(sdst + (uint32_t)((1 - p) * 64 * 32 * 2),
                  &W[(size_t)(n0 + srow) * K + k0 + 32 + sseg * 8]);
      WAIT_ASYNC(0x1);
    } else {
      WAIT_ASYNC(0x0);
    }
    __syncthreads();  // wt[p] ready for everyone

    // A ISA layout: lane<16 -> row li, K {0..7,16..23}; lane>=16 -> {8..15,24..31}
    V16 a[2];
#pragma unroll
    for (int rt = 0; rt < 2; ++rt) {
      const half_t* ap = A + (size_t)(m0 + rt * 16 + li) * K + k0 + (hi ? 8 : 0);
      a[rt].h[0] = *(const v8h*)ap;
      a[rt].h[1] = *(const v8h*)(ap + 16);
      if (more) __builtin_prefetch(ap + 32, 0, 0);  // global_prefetch_b8
    }
#pragma unroll
    for (int nn = 0; nn < 4; ++nn) {
      // B ISA layout: lane<16 -> col li, K {0..7,16..23}; lane>=16 -> {8..15,24..31}
      V16 b;
      const half_t* wp = &wt[p][(nn * 16 + li) * 32 + (hi ? 8 : 0)];
      b.h[0] = *(const v8h*)wp;
      b.h[1] = *(const v8h*)(wp + 16);
#pragma unroll
      for (int rt = 0; rt < 2; ++rt)
        acc[rt][nn] = wmma16(a[rt].v, b.v, acc[rt][nn]);
    }
    p ^= 1;
  }

#pragma unroll
  for (int rt = 0; rt < 2; ++rt) {
#pragma unroll
    for (int nn = 0; nn < 4; ++nn) {
      const int col = n0 + nn * 16 + li;
      float cs = scaleA;
      if constexpr (MODE == 1) cs *= colscale[col];
#pragma unroll
      for (int r = 0; r < 8; ++r) {
        const int rowg = m0 + rt * 16 + r + (hi ? 8 : 0);
        const float vv = acc[rt][nn][r] * cs;
        if constexpr (MODE == 1) {
          ((float*)Out)[(size_t)rowg * Nout + col] = vv;
        } else if constexpr (MODE == 2) {
          if (col < 64) {  // K head, row-major
            ((half_t*)Out)[(size_t)rowg * 64 + col] = (half_t)vv;
          } else {         // V head, transposed: vt[b][d][n]
            const int b = rowg >> 11, n = rowg & (N_ - 1);
            out2[((size_t)(b * 64 + (col - 64))) * N_ + n] = (half_t)vv;
          }
        } else {
          ((half_t*)Out)[(size_t)rowg * Nout + col] = (half_t)vv;
        }
      }
    }
  }
}

// ---------------------------------------------------------------------------
// Kernel 4: flash attention, 64-key chunks, async double-buffered K/V^T.
// grid = (N/16, B, 2); block = 256 = 8 waves; wave w -> head z*8+w; K/V shared
// across all heads (single KV head). kh: (B*N,64) row-major. vt: (B,64,N).
// ---------------------------------------------------------------------------
__global__ void __launch_bounds__(256)
flash_attn(const half_t* __restrict__ qh, const half_t* __restrict__ kh,
           const half_t* __restrict__ vth, half_t* __restrict__ oh) {
  __shared__ __align__(16) half_t kl[2][64 * 64];   // keys x d
  __shared__ __align__(16) half_t vtl[2][64 * 64];  // d x keys
  __shared__ __align__(16) half_t pb[8][16 * 64];   // per-wave P relayout
  const int tid = threadIdx.x;
  const int wv = tid >> 5, lane = tid & 31, li = lane & 15;
  const bool hi = lane >= 16;
  const int m0 = blockIdx.x * 16;
  const size_t bN = (size_t)blockIdx.y * N_;
  const int hc = (blockIdx.z * 8 + wv) * D_;

  // Q (pre-scaled by SCALE) in A-layout, two 32-wide d chunks, in VGPRs.
  V16 qa[2];
#pragma unroll
  for (int dc = 0; dc < 2; ++dc) {
    const half_t* qp =
        qh + (bN + m0 + li) * (size_t)(H_ * D_) + hc + dc * 32 + (hi ? 8 : 0);
    qa[dc].h[0] = *(const v8h*)qp;
    qa[dc].h[1] = *(const v8h*)(qp + 16);
  }

  float mrow[8], lrow[8];
  v8f o[4] = {};
#pragma unroll
  for (int r = 0; r < 8; ++r) { mrow[r] = -1e30f; lrow[r] = 0.f; }

  // staging: K chunk = contiguous 8KB (64 keys x 128B); V^T chunk = 64 rows of
  // 128B, row stride N_ halves. 512 x 16B units each; thread t -> {t, t+256}.
  const uint32_t klbase = lds_addr(&kl[0][0]);
  const uint32_t vtbase = lds_addr(&vtl[0][0]);
  const half_t* kbase = kh + (bN)*64;
  const half_t* vbase = vth + (size_t)blockIdx.y * 64 * N_;

  auto stage = [&](int j0, int buf) {
#pragma unroll
    for (int u8 = 0; u8 < 2; ++u8) {
      const int u = tid + u8 * 256;
      async_ld128(klbase + (uint32_t)(buf * 8192 + u * 16),
                  (const char*)(kbase + (size_t)j0 * 64) + u * 16);
      const int vr = u >> 3, vs = u & 7;
      async_ld128(vtbase + (uint32_t)(buf * 8192 + vr * 128 + vs * 16),
                  (const char*)(vbase + (size_t)vr * N_ + j0) + vs * 16);
    }
  };

  stage(0, 0);
  int p = 0;
  for (int j0 = 0; j0 < N_; j0 += 64) {
    __syncthreads();  // everyone done reading buffers 1-p
    const bool more = (j0 + 64) < N_;
    if (more) {
      stage(j0 + 64, 1 - p);
      WAIT_ASYNC(0x4);  // 4 newest = next chunk; current chunk has landed
    } else {
      WAIT_ASYNC(0x0);
    }
    __syncthreads();  // buffers p ready for everyone

    // S = Q*K^T : four 16x16 tiles (keys jj..jj+15), K-dim = d (2 steps of 32)
    v8f s[4] = {};
#pragma unroll
    for (int dc = 0; dc < 2; ++dc) {
#pragma unroll
      for (int jj = 0; jj < 4; ++jj) {
        V16 b;
        const half_t* kp = &kl[p][(jj * 16 + li) * 64 + dc * 32 + (hi ? 8 : 0)];
        b.h[0] = *(const v8h*)kp;
        b.h[1] = *(const v8h*)(kp + 16);
        s[jj] = wmma16(qa[dc].v, b.v, s[jj]);
      }
    }

    // online softmax over the 64 new columns: fold 4 tiles in VALU, then one
    // permlane16 butterfly per row for max and one for sum.
    half_t* pw = &pb[wv][0];
#pragma unroll
    for (int r = 0; r < 8; ++r) {
      const float a0 = s[0][r], a1 = s[1][r], a2 = s[2][r], a3 = s[3][r];
      const float mx = rmax16(fmaxf(fmaxf(a0, a1), fmaxf(a2, a3)));
      const float mnew = fmaxf(mrow[r], mx);
      const float corr = __expf(mrow[r] - mnew);
      const float p0 = __expf(a0 - mnew), p1 = __expf(a1 - mnew);
      const float p2 = __expf(a2 - mnew), p3 = __expf(a3 - mnew);
      lrow[r] = lrow[r] * corr + rsum16((p0 + p1) + (p2 + p3));
      mrow[r] = mnew;
#pragma unroll
      for (int c = 0; c < 4; ++c) o[c][r] = o[c][r] * corr;
      const int pr = r + (hi ? 8 : 0);
      pw[pr * 64 + li]      = (half_t)p0;
      pw[pr * 64 + 16 + li] = (half_t)p1;
      pw[pr * 64 + 32 + li] = (half_t)p2;
      pw[pr * 64 + 48 + li] = (half_t)p3;
    }
    // reload P in A-layout (two 16x32 A tiles), then O += P * V
#pragma unroll
    for (int kk = 0; kk < 2; ++kk) {
      V16 pa;
      const half_t* pp = pw + li * 64 + kk * 32 + (hi ? 8 : 0);
      pa.h[0] = *(const v8h*)pp;
      pa.h[1] = *(const v8h*)(pp + 16);
#pragma unroll
      for (int c = 0; c < 4; ++c) {
        V16 vb;
        const half_t* vp = &vtl[p][(c * 16 + li) * 64 + kk * 32 + (hi ? 8 : 0)];
        vb.h[0] = *(const v8h*)vp;
        vb.h[1] = *(const v8h*)(vp + 16);
        o[c] = wmma16(pa.v, vb.v, o[c]);
      }
    }
    p ^= 1;
  }

  // epilogue: O / l, write f16 to (B*N, H*D)
#pragma unroll
  for (int r = 0; r < 8; ++r) {
    const float inv = 1.f / lrow[r];
    const size_t rowg = bN + m0 + r + (hi ? 8 : 0);
#pragma unroll
    for (int c = 0; c < 4; ++c)
      oh[rowg * (size_t)(H_ * D_) + hc + c * 16 + li] = (half_t)(o[c][r] * inv);
  }
}

// ---------------------------------------------------------------------------
// launch
// ---------------------------------------------------------------------------
extern "C" void kernel_launch(void* const* d_in, const int* in_sizes, int n_in,
                              void* d_out, int out_size, void* d_ws, size_t ws_size,
                              hipStream_t stream) {
  (void)in_sizes; (void)n_in; (void)out_size; (void)ws_size;
  const float* x     = (const float*)d_in[0];
  const float* gamma = (const float*)d_in[1];
  const float* Wq    = (const float*)d_in[2];
  const float* Wkv   = (const float*)d_in[3];
  const float* Wo    = (const float*)d_in[4];
  const float* ls    = (const float*)d_in[5];
  float* out = (float*)d_out;

  char* ws = (char*)d_ws;
  size_t off = 0;
  auto carve = [&](size_t elems) -> half_t* {
    half_t* p = (half_t*)(ws + off);
    off += (elems * sizeof(half_t) + 255) & ~(size_t)255;
    return p;
  };
  half_t* xh   = carve((size_t)M_ * C_);      // normalized x, f16
  half_t* xf   = carve((size_t)M_ * C_);      // raw x, f16
  half_t* qhb  = carve((size_t)M_ * C_);      // Q (pre-scaled), f16
  half_t* khb  = carve((size_t)M_ * 64);      // K rows, f16
  half_t* vtb  = carve((size_t)B_ * 64 * N_); // V transposed (b,d,n), f16
  half_t* ohb  = carve((size_t)M_ * C_);      // attention output, f16
  half_t* wqh  = carve((size_t)C_ * C_);
  half_t* wkvh = carve((size_t)128 * C_);
  half_t* woh  = carve((size_t)C_ * C_);

  norm_cvt<<<M_, 256, 0, stream>>>(x, gamma, xh, xf);
  f2h<<<(C_ * C_ + 255) / 256, 256, 0, stream>>>(Wq, wqh, C_ * C_);
  f2h<<<(128 * C_ + 255) / 256, 256, 0, stream>>>(Wkv, wkvh, 128 * C_);
  f2h<<<(C_ * C_ + 255) / 256, 256, 0, stream>>>(Wo, woh, C_ * C_);

  // Q = xn @ Wq^T (fold in softmax scale)
  gemm_rt<0><<<dim3(C_ / 64, M_ / 256), 256, 0, stream>>>(
      xh, wqh, (void*)qhb, nullptr, nullptr, C_, C_, SCALE_);
  // [K|V] = x @ Wkv^T, V written transposed
  gemm_rt<2><<<dim3(2, M_ / 256), 256, 0, stream>>>(
      xf, wkvh, (void*)khb, vtb, nullptr, C_, 128, 1.0f);
  // fused flash attention
  flash_attn<<<dim3(N_ / 16, B_, 2), 256, 0, stream>>>(qhb, khb, vtb, ohb);
  // out = ls_scale * (attn_out @ Wo^T), f32
  gemm_rt<1><<<dim3(C_ / 64, M_ / 256), 256, 0, stream>>>(
      ohb, woh, (void*)out, nullptr, ls, C_, C_, 1.0f);
}
